// NormalizingFlow_3710851744111
// MI455X (gfx1250) — matrix operand order repaired and checked by reading, not compiled, not run on hardware
//
#include <hip/hip_runtime.h>

#define LAYERS  5
#define DIMX    128
#define SPLITD  64
#define HID     64
#define PITCH16 72   // bf16 halfword pitch: lane B128 reads map to disjoint 4-bank groups

typedef float  v8f   __attribute__((ext_vector_type(8)));
typedef __bf16 v8bf  __attribute__((ext_vector_type(8)));
typedef __bf16 v16bf __attribute__((ext_vector_type(16)));

static __device__ __forceinline__ v8f wmma_bf16(v16bf a, v16bf b, v8f c) {
    // D(16x16,f32) = A(16x32,bf16) x B(32x16,bf16) + C
    return __builtin_amdgcn_wmma_f32_16x16x32_bf16(
        /*neg_a=*/false, a, /*neg_b=*/false, b,
        /*c_mod=*/(short)0, c, /*reuse_a=*/false, /*reuse_b=*/false);
}

static __device__ __forceinline__ unsigned short bf16_bits(float v) {
    return __builtin_bit_cast(unsigned short, (__bf16)v);
}

static __device__ __forceinline__ float fast_tanh(float v) {
#if __has_builtin(__builtin_amdgcn_tanhf)
    return __builtin_amdgcn_tanhf(v);
#else
    float r;
    // native CDNA5 transcendental; v_nop covers the 1-op TRANS result hazard
    asm("v_tanh_f32 %0, %1\n\tv_nop" : "=v"(r) : "v"(v));
    return r;
#endif
}

// Packed B-fragment fetch: fragment(net,wsel,layer,kt,nt) is 32 lanes x 16 bf16.
static __device__ __forceinline__ v16bf load_frag(const unsigned short* __restrict__ wpk,
                                                  int net, int wsel, int layer,
                                                  int kt, int nt, int lane) {
    const unsigned short* p =
        wpk + ((size_t)(((((net * 3 + wsel) * LAYERS + layer) * 2 + kt) * 4 + nt) * 32 + lane)) * 16;
    return *(const v16bf*)(const void*)p;   // 32B aligned -> 2x global_load_b128
}

// A fragment from a row-major bf16 LDS tile: two contiguous 16B chunks per lane.
// halfwords j0..7  = K (kt*32 + group*8 + 0..7)       -> chunk at base
// halfwords j8..15 = K (kt*32 + 16 + group*8 + 0..7)  -> chunk at base+16
static __device__ __forceinline__ v16bf load_a(const unsigned short* h, int m, int group, int kt) {
    const int base = m * PITCH16 + kt * 32 + group * 8;
    const v8bf c0 = *(const v8bf*)(const void*)(h + base);
    const v8bf c1 = *(const v8bf*)(const void*)(h + base + 16);
    return __builtin_shufflevector(c0, c1, 0, 1, 2, 3, 4, 5, 6, 7,
                                   8, 9, 10, 11, 12, 13, 14, 15);
}

// One 3-layer MLP (64->64->64->64) for the 16 rows held by this wave.
static __device__ __forceinline__ void run_mlp(int net, int layer,
                                               const v16bf (&a0)[2],
                                               const unsigned short* __restrict__ wpk,
                                               const float* __restrict__ b1,
                                               const float* __restrict__ b2,
                                               const float* __restrict__ b3,
                                               unsigned short* __restrict__ hw,
                                               int lane, v8f (&out)[4]) {
    const int group = lane >> 4;
    const int m     = lane & 15;
    v16bf a[2];

    // ---- GEMM1: a0 x W1 + b1, leaky ReLU, stage to LDS as bf16 row-major ----
#pragma unroll
    for (int nt = 0; nt < 4; ++nt) {
        const float bv = b1[layer * HID + nt * 16 + m];
        v8f c;
#pragma unroll
        for (int r = 0; r < 8; ++r) c[r] = bv;
        c = wmma_bf16(a0[0], load_frag(wpk, net, 0, layer, 0, nt, lane), c);
        c = wmma_bf16(a0[1], load_frag(wpk, net, 0, layer, 1, nt, lane), c);
#pragma unroll
        for (int r = 0; r < 8; ++r) {
            float v = c[r];
            v = fmaxf(v, 0.01f * v);               // leaky ReLU (slope < 1)
            hw[(r + 8 * group) * PITCH16 + nt * 16 + m] = bf16_bits(v);
        }
    }
    a[0] = load_a(hw, m, group, 0);
    a[1] = load_a(hw, m, group, 1);

    // ---- GEMM2: h1 x W2 + b2, leaky ReLU, stage to LDS ----
#pragma unroll
    for (int nt = 0; nt < 4; ++nt) {
        const float bv = b2[layer * HID + nt * 16 + m];
        v8f c;
#pragma unroll
        for (int r = 0; r < 8; ++r) c[r] = bv;
        c = wmma_bf16(a[0], load_frag(wpk, net, 1, layer, 0, nt, lane), c);
        c = wmma_bf16(a[1], load_frag(wpk, net, 1, layer, 1, nt, lane), c);
#pragma unroll
        for (int r = 0; r < 8; ++r) {
            float v = c[r];
            v = fmaxf(v, 0.01f * v);
            hw[(r + 8 * group) * PITCH16 + nt * 16 + m] = bf16_bits(v);
        }
    }
    a[0] = load_a(hw, m, group, 0);
    a[1] = load_a(hw, m, group, 1);

    // ---- GEMM3: h2 x W3 + b3 -> out (no activation) ----
#pragma unroll
    for (int nt = 0; nt < 4; ++nt) {
        const float bv = b3[layer * HID + nt * 16 + m];
        v8f c;
#pragma unroll
        for (int r = 0; r < 8; ++r) c[r] = bv;
        c = wmma_bf16(a[0], load_frag(wpk, net, 2, layer, 0, nt, lane), c);
        c = wmma_bf16(a[1], load_frag(wpk, net, 2, layer, 1, nt, lane), c);
        out[nt] = c;
    }
}

// Pre-swizzle all 30 weight matrices (f32 -> bf16) into WMMA B-fragment order.
__global__ void flow_pack_weights(const float* __restrict__ sW1, const float* __restrict__ sW2,
                                  const float* __restrict__ sW3, const float* __restrict__ tW1,
                                  const float* __restrict__ tW2, const float* __restrict__ tW3,
                                  unsigned short* __restrict__ wpk) {
    const int gid = blockIdx.x * 256 + threadIdx.x;
    if (gid >= 30 * 4096) return;
    const int j    = gid & 15;
    const int lane = (gid >> 4) & 31;
    const int nt   = (gid >> 9) & 3;
    const int kt   = (gid >> 11) & 1;
    const int mi   = gid >> 12;               // (net*3 + wsel)*5 + layer
    const int net = mi / 15, rem = mi % 15, wsel = rem / 5, layer = rem % 5;
    // B 32x16 layout: K = kt*32 + (j<8 ? (lane/16)*8 + j : 16 + (lane/16)*8 + j-8) ; N = nt*16 + lane%16
    const int k = kt * 32 + (j < 8 ? (lane >> 4) * 8 + j : 16 + (lane >> 4) * 8 + (j - 8));
    const int n = nt * 16 + (lane & 15);
    const float* W = (net == 0) ? (wsel == 0 ? sW1 : (wsel == 1 ? sW2 : sW3))
                                : (wsel == 0 ? tW1 : (wsel == 1 ? tW2 : tW3));
    wpk[gid] = bf16_bits(W[layer * 4096 + k * 64 + n]);
}

__global__ void __launch_bounds__(256, 1)
flow_main(const float* __restrict__ x,
          const float* __restrict__ sb1, const float* __restrict__ sb2,
          const float* __restrict__ sb3, const float* __restrict__ tb1,
          const float* __restrict__ tb2, const float* __restrict__ tb3,
          const unsigned short* __restrict__ wpk,
          float* __restrict__ z, float* __restrict__ logdet) {
    __shared__ unsigned short hb[8][16 * PITCH16];   // wave-private bf16 activation staging

    const int wave    = threadIdx.x >> 5;
    const int lane    = threadIdx.x & 31;
    const int rowBase = blockIdx.x * 128 + wave * 16;
    const int group   = lane >> 4;
    const int m       = lane & 15;
    unsigned short* hw = hb[wave];

    // ---- coalesced load of x0 tile (16 rows x 64 f32): copy to z[:, :64],
    //      convert to bf16 and stage row-major in LDS (8B packed stores) ----
#pragma unroll
    for (int t = 0; t < 8; ++t) {
        const int fi = t * 32 + lane;          // 256 float4 slots, 16 per row
        const int rr = fi >> 4;
        const int c4 = (fi & 15) << 2;
        const float4 v = *(const float4*)(x + (size_t)(rowBase + rr) * DIMX + c4);
        *(float4*)(z + (size_t)(rowBase + rr) * DIMX + c4) = v;
        const unsigned long long pk =
            (unsigned long long)bf16_bits(v.x) |
            ((unsigned long long)bf16_bits(v.y) << 16) |
            ((unsigned long long)bf16_bits(v.z) << 32) |
            ((unsigned long long)bf16_bits(v.w) << 48);
        *(unsigned long long*)(void*)&hw[rr * PITCH16 + c4] = pk;
    }

    // A fragments of x0: built once (4x ds_load_b128), reused by all 10 MLPs
    v16bf a0[2];
    a0[0] = load_a(hw, m, group, 0);
    a0[1] = load_a(hw, m, group, 1);

    // x1 tile in C-fragment layout (lane holds col n = nt*16+m for rows r+8*group)
    float x1r[4][8];
#pragma unroll
    for (int nt = 0; nt < 4; ++nt)
#pragma unroll
        for (int r = 0; r < 8; ++r)
            x1r[nt][r] = x[(size_t)(rowBase + r + 8 * group) * DIMX + SPLITD + nt * 16 + m];

    float ld[8];
#pragma unroll
    for (int r = 0; r < 8; ++r) ld[r] = 0.f;

    for (int l = 0; l < LAYERS; ++l) {
        v8f s4[4], t4[4];
        run_mlp(0, l, a0, wpk, sb1, sb2, sb3, hw, lane, s4);
        run_mlp(1, l, a0, wpk, tb1, tb2, tb3, hw, lane, t4);
#pragma unroll
        for (int nt = 0; nt < 4; ++nt)
#pragma unroll
            for (int r = 0; r < 8; ++r) {
                const float sv = fast_tanh(s4[nt][r]);   // native v_tanh_f32
                ld[r] += sv;
                x1r[nt][r] = fmaf(x1r[nt][r], __expf(sv), t4[nt][r]);
            }
    }

    // ---- store x1 -> z[:, 64:] ----
#pragma unroll
    for (int nt = 0; nt < 4; ++nt)
#pragma unroll
        for (int r = 0; r < 8; ++r)
            z[(size_t)(rowBase + r + 8 * group) * DIMX + SPLITD + nt * 16 + m] = x1r[nt][r];

    // ---- log_det: reduce over the 16 lanes of each half-wave ----
#pragma unroll
    for (int r = 0; r < 8; ++r) {
        float v = ld[r];
        v += __shfl_xor(v, 1);
        v += __shfl_xor(v, 2);
        v += __shfl_xor(v, 4);
        v += __shfl_xor(v, 8);
        if ((lane & 15) == 0) logdet[rowBase + r + 8 * group] = v;
    }
}

extern "C" void kernel_launch(void* const* d_in, const int* in_sizes, int n_in,
                              void* d_out, int out_size, void* d_ws, size_t ws_size,
                              hipStream_t stream) {
    const float* x   = (const float*)d_in[0];
    const float* sW1 = (const float*)d_in[1];
    const float* sb1 = (const float*)d_in[2];
    const float* sW2 = (const float*)d_in[3];
    const float* sb2 = (const float*)d_in[4];
    const float* sW3 = (const float*)d_in[5];
    const float* sb3 = (const float*)d_in[6];
    const float* tW1 = (const float*)d_in[7];
    const float* tb1 = (const float*)d_in[8];
    const float* tW2 = (const float*)d_in[9];
    const float* tb2 = (const float*)d_in[10];
    const float* tW3 = (const float*)d_in[11];
    const float* tb3 = (const float*)d_in[12];

    const int B = in_sizes[0] / DIMX;             // 262144
    unsigned short* wpk = (unsigned short*)d_ws;  // 30*4096 bf16 = 240 KB

    flow_pack_weights<<<(30 * 4096 + 255) / 256, 256, 0, stream>>>(
        sW1, sW2, sW3, tW1, tW2, tW3, wpk);

    float* z    = (float*)d_out;
    float* ldet = z + (size_t)B * DIMX;
    flow_main<<<B / 128, 256, 0, stream>>>(x, sb1, sb2, sb3, tb1, tb2, tb3, wpk, z, ldet);
}
